// Precept_85779086835960
// MI455X (gfx1250) — matrix operand order repaired and checked by reading, not compiled, not run on hardware
//
#include <hip/hip_runtime.h>

// ---------------------------------------------------------------------------
// CDNA5 (gfx1250) wave32 WMMA helpers.  ISA: CDNA5 ISA layouts.
// ---------------------------------------------------------------------------
typedef _Float16 v16h __attribute__((ext_vector_type(16)));
typedef float    v8f  __attribute__((ext_vector_type(8)));
typedef const v16h __attribute__((address_space(1))) *gbl_v16h;

__device__ inline v8f wmma_f16(v16h a, v16h b, v8f c) {
  // D = A(16x32 f16) * B(32x16 f16) + C(16x16 f32)
  return __builtin_amdgcn_wmma_f32_16x16x32_f16(false, a, false, b, (short)0, c,
                                                false, false);
}

// A-operand gather from LDS f16 array (row-major, padded stride).
// 16-bit A 16x32 layout: lane<16 -> K {kb..kb+15} with kb=0;
// lane>=16 -> kb=16 (contiguous, matches packed B).
__device__ inline v16h lda_frag(const _Float16* base, int stride, int kbase) {
  int lane = threadIdx.x & 31;
  int m = lane & 15;
  int kb = (lane >> 4) << 4;
  const unsigned* r0 = (const unsigned*)(base + m * stride + kbase + kb);
  const unsigned* r1 = (const unsigned*)(base + m * stride + kbase + kb + 8);
  union { v16h h; unsigned u[8]; } t;
#pragma unroll
  for (int j = 0; j < 4; ++j) { t.u[j] = r0[j]; t.u[4 + j] = r1[j]; }
  return t.h;
}

// Same gather but from an f32 LDS array (convert on read).
__device__ inline v16h lda_frag_f32(const float* base, int stride, int kbase) {
  int lane = threadIdx.x & 31;
  int m = lane & 15;
  int kb = (lane >> 4) << 4;
  const float* r0 = base + m * stride + kbase + kb;
  v16h u;
#pragma unroll
  for (int j = 0; j < 8; ++j) u[j] = (_Float16)r0[j];
#pragma unroll
  for (int j = 0; j < 8; ++j) u[8 + j] = (_Float16)r0[8 + j];
  return u;
}

// B-operand from pre-packed global weights: force addrspace(1) so this lowers
// to global_load_b128 (flat loads would also tie up DScnt).
__device__ inline v16h ldb_packed(const _Float16* wp, int tile) {
  const _Float16* p = wp + ((size_t)tile * 32 + (threadIdx.x & 31)) * 16;
  return *(gbl_v16h)(unsigned long long)p;
}

// B-operand for K=16 (padded to 32) built from an f32 LDS matrix h[16][stride]:
// lanes 0-15 carry K=0..15 (real rows), lanes 16-31 carry K=16..31 (zeros).
__device__ inline v16h ldb_k16_f32(const float* h, int stride, int hbase) {
  int lane = threadIdx.x & 31;
  v16h u;
  if (lane < 16) {
#pragma unroll
    for (int j = 0; j < 16; ++j) u[j] = (_Float16)h[j * stride + hbase + lane];
  } else {
#pragma unroll
    for (int j = 0; j < 16; ++j) u[j] = (_Float16)0.f;
  }
  return u;
}

// Store C/D fragment (16x16 f32) to LDS: VGPR r -> row r + 8*(lane>=16), col lane&15.
__device__ inline void stc_f32(float* dst, int stride, int hbase, v8f c) {
  int lane = threadIdx.x & 31;
  int n = lane & 15, moff = (lane >> 4) << 3;
#pragma unroll
  for (int r = 0; r < 8; ++r) dst[(moff + r) * stride + hbase + n] = c[r];
}

// Hardware reciprocal (single v_rcp_f32; avoids the precise-division
// v_div_scale/fixup expansion seen in round-4 asm).
__device__ inline float fast_rcp(float x) { return __builtin_amdgcn_rcpf(x); }

__device__ inline float sigm(float x) {
  return fast_rcp(1.f + __expf(-x));
}

// Branchless fast tanh: 1 - 2/(1+e^{2x}).  Exact at both saturated ends,
// one v_exp_f32 + one v_rcp_f32, no EXEC divergence.
__device__ inline float fast_tanh(float x) {
  return 1.f - 2.f * fast_rcp(1.f + __expf(2.f * x));
}

// ---------------------------------------------------------------------------
// Async global->LDS copy (ASYNCcnt-tracked DMA path, ISA 15.18.3 op 96-98).
// LDS byte address = low 32 bits of the generic pointer (ISA 10.2).
// ---------------------------------------------------------------------------
__device__ inline void async_g2l_b128(const void* gsrc, void* ldst) {
  unsigned lds = (unsigned)(unsigned long long)ldst;
  asm volatile("global_load_async_to_lds_b128 %0, %1, off" ::"v"(lds),
               "v"((unsigned long long)gsrc)
               : "memory");
}
__device__ inline void wait_asynccnt0() {
  asm volatile("s_wait_asynccnt 0" ::: "memory");
}

// ---------------------------------------------------------------------------
// Weight pre-pack: fp32 [R,C] (optionally transposed) -> f16 in WMMA-B layout.
// ---------------------------------------------------------------------------
__global__ void pack_w_kernel(const float* __restrict__ src,
                              _Float16* __restrict__ dst, int R, int C,
                              int trans) {
  int K = trans ? C : R;
  int N = trans ? R : C;
  int htiles = N >> 4;
  int tile = blockIdx.x;
  int kt = tile / htiles, ht = tile - kt * htiles;
  int lane = threadIdx.x;
  int kb = kt * 32 + ((lane >> 4) << 4);
  int n = ht * 16 + (lane & 15);
  _Float16* q = dst + ((size_t)tile * 32 + lane) * 16;
#pragma unroll
  for (int j = 0; j < 16; ++j) {
    int k = kb + j;
    float v = 0.f;
    if (k < K) v = trans ? src[(size_t)n * C + k] : src[(size_t)k * C + n];
    q[j] = (_Float16)v;
  }
}

// ---------------------------------------------------------------------------
// GAT attention: f1/f2 = h@a1, h@a2; e = lrelu(f1_i+f2_j) masked; softmax per
// row -> attb (f16, cols 16..31 pre-zeroed).
// ---------------------------------------------------------------------------
__device__ inline void gat_attention(const float* hm, int hstride,
                                     const float* __restrict__ avec,
                                     const int* adjs, _Float16* attb,
                                     int astride, float* pf1, float* pf2,
                                     float* f1s, float* f2s) {
  int tid = threadIdx.x;
  int i = tid & 15, c = tid >> 4;
  const float* a1 = avec;
  const float* a2 = avec + 128;
  float s1 = 0.f, s2 = 0.f;
#pragma unroll
  for (int k = 0; k < 16; ++k) {
    float hv = hm[i * hstride + c * 16 + k];
    s1 += hv * a1[c * 16 + k];
    s2 += hv * a2[c * 16 + k];
  }
  pf1[i * 8 + c] = s1;
  pf2[i * 8 + c] = s2;
  __syncthreads();
  if (tid < 32) {
    int ii = tid & 15;
    float s = 0.f;
    if (tid < 16) { for (int cc = 0; cc < 8; ++cc) s += pf1[ii * 8 + cc]; f1s[ii] = s; }
    else          { for (int cc = 0; cc < 8; ++cc) s += pf2[ii * 8 + cc]; f2s[ii] = s; }
  }
  __syncthreads();
  if (tid < 16) {
    float e[16], mx = -3.0e38f;
#pragma unroll
    for (int j = 0; j < 16; ++j) {
      float v = f1s[tid] + f2s[j];
      v = (v >= 0.f) ? v : 0.2f * v;          // leaky_relu slope 0.2
      if (adjs[tid * 16 + j] <= 0) v = -1.0e9f;
      e[j] = v;
      mx = fmaxf(mx, v);
    }
    float ss = 0.f;
#pragma unroll
    for (int j = 0; j < 16; ++j) { float ex = __expf(e[j] - mx); e[j] = ex; ss += ex; }
    float inv = fast_rcp(ss);
#pragma unroll
    for (int j = 0; j < 16; ++j) attb[tid * astride + j] = (_Float16)(e[j] * inv);
  }
  __syncthreads();
}

// Row layernorm stats over 128 cols of h[16][stride] (+optional bias).
template <bool BIAS>
__device__ inline void row_stats(const float* h, int stride,
                                 const float* __restrict__ bias, float* pf1,
                                 float* pf2, float* mu, float* rstd) {
  int tid = threadIdx.x;
  int i = tid & 15, c = tid >> 4;
  float s = 0.f, s2 = 0.f;
#pragma unroll
  for (int k = 0; k < 16; ++k) {
    float v = h[i * stride + c * 16 + k] + (BIAS ? bias[c * 16 + k] : 0.f);
    s += v;
    s2 += v * v;
  }
  pf1[i * 8 + c] = s;
  pf2[i * 8 + c] = s2;
  __syncthreads();
  if (tid < 16) {
    float a = 0.f, q = 0.f;
    for (int cc = 0; cc < 8; ++cc) { a += pf1[tid * 8 + cc]; q += pf2[tid * 8 + cc]; }
    float m = a * (1.f / 128.f);
    float var = q * (1.f / 128.f) - m * m;
    mu[tid] = m;
    rstd[tid] = rsqrtf(fmaxf(var, 0.f) + 1e-5f);
  }
  __syncthreads();
}

// h = t @ W (K=128): hoist the 4 A-fragments, reuse for both H tiles.
__device__ inline void gat_matmul128(_Float16 (*hb)[136],
                                     const _Float16* __restrict__ Wp,
                                     float (*hm)[132]) {
  int wid = threadIdx.x >> 5;
  v16h af[4];
#pragma unroll
  for (int kc = 0; kc < 4; ++kc) af[kc] = lda_frag(&hb[0][0], 136, kc * 32);
#pragma unroll
  for (int t = 0; t < 2; ++t) {
    int ht = wid * 2 + t;
    v8f acc = {};
#pragma unroll
    for (int kc = 0; kc < 4; ++kc)
      acc = wmma_f16(af[kc], ldb_packed(Wp, kc * 8 + ht), acc);
    stc_f32(&hm[0][0], 132, ht * 16, acc);
  }
}

// attention + apply + (residual) + tanh -> hs (f32) and hb (f16)
template <bool RES>
__device__ inline void gat_att_apply(float (*hs)[132], float (*hm)[132],
                                     _Float16 (*hb)[136], _Float16 (*attb)[48],
                                     const float* __restrict__ avec,
                                     const int* adjs, float* pf1, float* pf2,
                                     float* f1s, float* f2s) {
  __syncthreads();
  gat_attention(&hm[0][0], 132, avec, adjs, &attb[0][0], 48, pf1, pf2, f1s, f2s);
  int tid = threadIdx.x, wid = tid >> 5;
  v16h aa = lda_frag(&attb[0][0], 48, 0);
#pragma unroll
  for (int t = 0; t < 2; ++t) {
    int ht = wid * 2 + t;
    v8f acc = {};
    acc = wmma_f16(aa, ldb_k16_f32(&hm[0][0], 132, ht * 16), acc);
    int lane = tid & 31, n = lane & 15, moff = (lane >> 4) << 3;
#pragma unroll
    for (int r = 0; r < 8; ++r) {
      int m = moff + r, col = ht * 16 + n;
      float v = acc[r] + (RES ? hs[m][col] : 0.f);
      v = fast_tanh(v);
      hs[m][col] = v;
      hb[m][col] = (_Float16)v;
    }
  }
  __syncthreads();
}

// ---------------------------------------------------------------------------
// job_enc: 4-layer GAT stack over one 16-node graph per block (4 waves).
// x and adj staged into LDS via async DMA; layer-1 A comes straight from the
// f32 staging buffer (no f16 round-trip).
// ---------------------------------------------------------------------------
__global__ __launch_bounds__(128) void job_enc_kernel(
    const float* __restrict__ xin, const int* __restrict__ adj,
    const _Float16* __restrict__ W1p, const _Float16* __restrict__ W2p,
    const _Float16* __restrict__ W3p, const _Float16* __restrict__ W4p,
    const float* __restrict__ avecs, float* __restrict__ out_emb) {
  __shared__ float xs32[16][40];     // x [16][32] f32, async-staged
  __shared__ _Float16 hb[16][136];   // current t (f16, A operand)
  __shared__ float hs[16][132];      // current t (f32, residual + mean)
  __shared__ float hm[16][132];      // h = t @ W
  __shared__ _Float16 attb[16][48];  // attention matrix, K padded to 32
  __shared__ float pf1[128], pf2[128], f1s[16], f2s[16];
  __shared__ __align__(16) int adjs[256];
  int tid = threadIdx.x;
  int g = blockIdx.x;
  int wid = tid >> 5;
  __builtin_prefetch(W2p, 0, 3);  // global_prefetch_b8: warm packed weights

  {  // async stage: x = 128 x 16B chunks, adj = 64 x 16B chunks
    int i = tid >> 3, k4 = tid & 7;
    async_g2l_b128(xin + (size_t)g * 512 + i * 32 + k4 * 4, &xs32[i][k4 * 4]);
    if (tid < 64) async_g2l_b128(adj + (size_t)g * 256 + tid * 4, &adjs[tid * 4]);
  }
  for (int idx = tid; idx < 256; idx += 128)
    attb[idx >> 4][16 + (idx & 15)] = (_Float16)0.f;  // zero K pad
  wait_asynccnt0();
  __syncthreads();

  {  // layer 1: h = x @ W1 (K=32), A-frag shared across both tiles
    v16h xa = lda_frag_f32(&xs32[0][0], 40, 0);
#pragma unroll
    for (int t = 0; t < 2; ++t) {
      int ht = wid * 2 + t;
      v8f acc = {};
      acc = wmma_f16(xa, ldb_packed(W1p, ht), acc);
      stc_f32(&hm[0][0], 132, ht * 16, acc);
    }
  }
  gat_att_apply<false>(hs, hm, hb, attb, avecs + 0, adjs, pf1, pf2, f1s, f2s);
  gat_matmul128(hb, W2p, hm);
  gat_att_apply<true>(hs, hm, hb, attb, avecs + 256, adjs, pf1, pf2, f1s, f2s);
  gat_matmul128(hb, W3p, hm);
  gat_att_apply<true>(hs, hm, hb, attb, avecs + 512, adjs, pf1, pf2, f1s, f2s);
  gat_matmul128(hb, W4p, hm);
  gat_att_apply<true>(hs, hm, hb, attb, avecs + 768, adjs, pf1, pf2, f1s, f2s);

  float s = 0.f;
#pragma unroll
  for (int i = 0; i < 16; ++i) s += hs[i][tid];
  out_emb[(size_t)g * 128 + tid] = s * (1.f / 16.f);
}

// ---------------------------------------------------------------------------
// LSTM: 16 independent sequences per block.  x(t+1) is async-DMA'd into LDS
// while step t's WMMA chain runs; A-fragments hoisted to registers and reused
// across all 8 gate tiles of each wave.
// ---------------------------------------------------------------------------
__global__ __launch_bounds__(128) void lstm_kernel(
    const float* __restrict__ xall, const _Float16* __restrict__ Wihp,
    const _Float16* __restrict__ Whhp, const float* __restrict__ bias,
    float* __restrict__ hout) {
  __shared__ float xf32[16][132];  // x_t staging (async target)
  __shared__ _Float16 hb[16][136];
  __shared__ float hcur[16][128];
  __shared__ float ccur[16][128];
  __shared__ float g[16][520];
  int tid = threadIdx.x;
  int wid = tid >> 5, lane = tid & 31;
  int seq0 = blockIdx.x * 16;
  __builtin_prefetch(Whhp, 0, 3);
  for (int idx = tid; idx < 2048; idx += 128) {
    int i = idx >> 7, k = idx & 127;
    hcur[i][k] = 0.f;
    ccur[i][k] = 0.f;
    hb[i][k] = (_Float16)0.f;
  }
  // kick off async fetch of x(0): 512 x 16B chunks, 4 per thread
#pragma unroll
  for (int cc = 0; cc < 4; ++cc) {
    int c = tid + cc * 128;
    int i = c >> 5, k4 = c & 31;
    async_g2l_b128(xall + ((size_t)(seq0 + i) * 64 + 0) * 128 + k4 * 4,
                   &xf32[i][k4 * 4]);
  }
  for (int t = 0; t < 64; ++t) {
    wait_asynccnt0();
    __syncthreads();  // xf32 ready, hb from previous step ready
    v16h xf[4], hf[4];
#pragma unroll
    for (int kc = 0; kc < 4; ++kc)
      xf[kc] = lda_frag_f32(&xf32[0][0], 132, kc * 32);
    __syncthreads();  // all waves done reading xf32
    if (t + 1 < 64) {  // overlap next-step fetch with the WMMA chain
#pragma unroll
      for (int cc = 0; cc < 4; ++cc) {
        int c = tid + cc * 128;
        int i = c >> 5, k4 = c & 31;
        async_g2l_b128(xall + ((size_t)(seq0 + i) * 64 + (t + 1)) * 128 + k4 * 4,
                       &xf32[i][k4 * 4]);
      }
    }
#pragma unroll
    for (int kc = 0; kc < 4; ++kc)
      hf[kc] = lda_frag(&hb[0][0], 136, kc * 32);
#pragma unroll 2
    for (int tt = 0; tt < 8; ++tt) {  // wave owns 8 of 32 gate tiles
      int nt = wid * 8 + tt;
      v8f acc = {};
#pragma unroll
      for (int kc = 0; kc < 4; ++kc) {
        acc = wmma_f16(xf[kc], ldb_packed(Wihp, kc * 32 + nt), acc);
        acc = wmma_f16(hf[kc], ldb_packed(Whhp, kc * 32 + nt), acc);
      }
      int n = lane & 15, moff = (lane >> 4) << 3;
#pragma unroll
      for (int r = 0; r < 8; ++r)
        g[moff + r][nt * 16 + n] = acc[r] + bias[nt * 16 + n];
    }
    __syncthreads();
    for (int idx = tid; idx < 2048; idx += 128) {
      int i = idx >> 7, k = idx & 127;
      float ig = g[i][k], fg = g[i][128 + k], gg = g[i][256 + k],
            og = g[i][384 + k];
      float c = sigm(fg) * ccur[i][k] + sigm(ig) * fast_tanh(gg);
      float h = sigm(og) * fast_tanh(c);
      ccur[i][k] = c;
      hcur[i][k] = h;
      hb[i][k] = (_Float16)h;
    }
    __syncthreads();
  }
  for (int idx = tid; idx < 2048; idx += 128) {
    int i = idx >> 7, k = idx & 127;
    hout[(size_t)(seq0 + i) * 128 + k] = hcur[i][k];
  }
}

// ---------------------------------------------------------------------------
// Final: machine embedding + concat projection + 4 layernormed GAT layers +
// mean over nodes.  One block per batch element.
// ---------------------------------------------------------------------------
__global__ __launch_bounds__(128) void machine_kernel(
    const float* __restrict__ mf, const int* __restrict__ ma,
    const _Float16* __restrict__ mfeWp, const float* __restrict__ mfe_b,
    const float* __restrict__ n2g, const float* __restrict__ n2b,
    const _Float16* __restrict__ meWp, const float* __restrict__ me_b,
    const float* __restrict__ n3g, const float* __restrict__ n3b,
    const _Float16* __restrict__ mgWp, const float* __restrict__ mga,
    const float* __restrict__ mng, const float* __restrict__ mnb,
    const float* __restrict__ jenc, const float* __restrict__ jm,
    float* __restrict__ out) {
  __shared__ _Float16 hb[16][136];
  __shared__ _Float16 cb[16][280];  // concat(mfh, ctx) as f16 A operand (K=256)
  __shared__ float hs[16][132];
  __shared__ float hm[16][132];
  __shared__ _Float16 attb[16][48];
  __shared__ float pf1[128], pf2[128], f1s[16], f2s[16], mu[16], rstd[16];
  __shared__ int adjs[256];
  int tid = threadIdx.x;
  int b = blockIdx.x;
  int wid = tid >> 5;

  for (int idx = tid; idx < 512; idx += 128) {  // MF=16, K padded to 32
    int i = idx >> 5, k = idx & 31;
    hb[i][k] = (_Float16)((k < 16) ? mf[(size_t)b * 256 + i * 16 + k] : 0.f);
  }
  for (int idx = tid; idx < 256; idx += 128) adjs[idx] = ma[(size_t)b * 256 + idx];
  for (int idx = tid; idx < 256; idx += 128)
    attb[idx >> 4][16 + (idx & 15)] = (_Float16)0.f;
  __syncthreads();

  // mfh = tanh(LN(mf @ mfe_W + mfe_b)) -> cb[:, 0:128]
  {
    v16h xa = lda_frag(&hb[0][0], 136, 0);
#pragma unroll
    for (int t = 0; t < 2; ++t) {
      int ht = wid * 2 + t;
      v8f acc = {};
      acc = wmma_f16(xa, ldb_packed(mfeWp, ht), acc);
      stc_f32(&hm[0][0], 132, ht * 16, acc);
    }
  }
  __syncthreads();
  row_stats<true>(&hm[0][0], 132, mfe_b, pf1, pf2, mu, rstd);
  for (int idx = tid; idx < 2048; idx += 128) {
    int i = idx >> 7, k = idx & 127;
    float v = hm[i][k] + mfe_b[k];
    v = (v - mu[i]) * rstd[i] * n2g[k] + n2b[k];
    cb[i][k] = (_Float16)fast_tanh(v);
  }
  // ctx: rows 0..7 <- j_encoded[b, s] = lstm_out[s*64+b]; rows 8..15 <- jm[b,m]
  for (int idx = tid; idx < 2048; idx += 128) {
    int i = idx >> 7, k = idx & 127;
    const float* src = (i < 8) ? (jenc + ((size_t)i * 64 + b) * 128)
                               : (jm + ((size_t)b * 8 + (i - 8)) * 128);
    cb[i][128 + k] = (_Float16)src[k];
  }
  __syncthreads();

  // nodes = tanh(LN(cb @ me_W + me_b)), K=256; hoist the 8 A-frags
  {
    v16h af[8];
#pragma unroll
    for (int kc = 0; kc < 8; ++kc)
      af[kc] = lda_frag(&cb[0][0], 280, kc * 32);
#pragma unroll
    for (int t = 0; t < 2; ++t) {
      int ht = wid * 2 + t;
      v8f acc = {};
#pragma unroll
      for (int kc = 0; kc < 8; ++kc)
        acc = wmma_f16(af[kc], ldb_packed(meWp, kc * 8 + ht), acc);
      stc_f32(&hm[0][0], 132, ht * 16, acc);
    }
  }
  __syncthreads();
  row_stats<true>(&hm[0][0], 132, me_b, pf1, pf2, mu, rstd);
  for (int idx = tid; idx < 2048; idx += 128) {
    int i = idx >> 7, k = idx & 127;
    float v = hm[i][k] + me_b[k];
    v = (v - mu[i]) * rstd[i] * n3g[k] + n3b[k];
    v = fast_tanh(v);
    hs[i][k] = v;
    hb[i][k] = (_Float16)v;
  }
  __syncthreads();

  // 4 GAT layers with layernorm + residual(l>0) + tanh.
#pragma unroll
  for (int l = 0; l < 4; ++l) {
    const _Float16* __restrict__ Wl = mgWp + (size_t)l * 16384;
    gat_matmul128(hb, Wl, hm);
    __syncthreads();
    gat_attention(&hm[0][0], 132, mga + l * 256, adjs, &attb[0][0], 48, pf1,
                  pf2, f1s, f2s);
    {
      v16h aa = lda_frag(&attb[0][0], 48, 0);
#pragma unroll
      for (int t = 0; t < 2; ++t) {  // gat out overwrites own columns of hm
        int ht = wid * 2 + t;
        v8f acc = {};
        acc = wmma_f16(aa, ldb_k16_f32(&hm[0][0], 132, ht * 16), acc);
        stc_f32(&hm[0][0], 132, ht * 16, acc);
      }
    }
    __syncthreads();
    row_stats<false>(&hm[0][0], 132, nullptr, pf1, pf2, mu, rstd);
    for (int idx = tid; idx < 2048; idx += 128) {
      int i = idx >> 7, k = idx & 127;
      float v = (hm[i][k] - mu[i]) * rstd[i] * mng[l * 128 + k] + mnb[l * 128 + k];
      if (l > 0) v += hs[i][k];
      v = fast_tanh(v);
      hs[i][k] = v;
      hb[i][k] = (_Float16)v;
    }
    __syncthreads();
  }
  float s = 0.f;
#pragma unroll
  for (int i = 0; i < 16; ++i) s += hs[i][tid];
  out[(size_t)b * 128 + tid] = s * (1.f / 16.f);
}

// ---------------------------------------------------------------------------
extern "C" void kernel_launch(void* const* d_in, const int* in_sizes, int n_in,
                              void* d_out, int out_size, void* d_ws,
                              size_t ws_size, hipStream_t stream) {
  (void)in_sizes; (void)n_in; (void)out_size; (void)ws_size;
  const float* j_stages_f  = (const float*)d_in[0];
  const int*   j_stages_a  = (const int*)d_in[1];
  const float* j_machine_f = (const float*)d_in[2];
  const int*   j_machine_a = (const int*)d_in[3];
  const float* machine_f   = (const float*)d_in[4];
  const int*   machine_a   = (const int*)d_in[5];
  const float* jg1_W    = (const float*)d_in[6];
  const float* jgW      = (const float*)d_in[7];
  const float* jga      = (const float*)d_in[8];
  const float* lstm_Wih = (const float*)d_in[9];
  const float* lstm_Whh = (const float*)d_in[10];
  const float* lstm_b   = (const float*)d_in[11];
  const float* mfe_W = (const float*)d_in[12];
  const float* mfe_b = (const float*)d_in[13];
  const float* n2_g  = (const float*)d_in[14];
  const float* n2_b  = (const float*)d_in[15];
  const float* me_W  = (const float*)d_in[16];
  const float* me_b  = (const float*)d_in[17];
  const float* n3_g  = (const float*)d_in[18];
  const float* n3_b  = (const float*)d_in[19];
  const float* mgW   = (const float*)d_in[20];
  const float* mga   = (const float*)d_in[21];
  const float* mng   = (const float*)d_in[22];
  const float* mnb   = (const float*)d_in[23];

  char* ws = (char*)d_ws;
  size_t off = 0;
  auto take = [&](size_t bytes) {
    char* p = ws + off;
    off = (off + bytes + 255) & ~(size_t)255;
    return p;
  };
  _Float16* W1p   = (_Float16*)take((size_t)1 * 8 * 512 * 2);
  _Float16* W2p   = (_Float16*)take((size_t)4 * 8 * 512 * 2);
  _Float16* W3p   = (_Float16*)take((size_t)4 * 8 * 512 * 2);
  _Float16* W4p   = (_Float16*)take((size_t)4 * 8 * 512 * 2);
  _Float16* Wihp  = (_Float16*)take((size_t)4 * 32 * 512 * 2);
  _Float16* Whhp  = (_Float16*)take((size_t)4 * 32 * 512 * 2);
  _Float16* mfeWp = (_Float16*)take((size_t)1 * 8 * 512 * 2);
  _Float16* meWp  = (_Float16*)take((size_t)8 * 8 * 512 * 2);
  _Float16* mgWp  = (_Float16*)take((size_t)4 * 4 * 8 * 512 * 2);
  float* stage_emb = (float*)take((size_t)32768 * 128 * 4);
  float* jmbuf     = (float*)take((size_t)512 * 128 * 4);
  float* jencbuf   = (float*)take((size_t)512 * 128 * 4);

  // Weight packing into WMMA-B layout (tiny; hot in L2 afterwards).
  pack_w_kernel<<<8, 32, 0, stream>>>(jg1_W, W1p, 32, 128, 0);
  pack_w_kernel<<<32, 32, 0, stream>>>(jgW + 0 * 16384, W2p, 128, 128, 0);
  pack_w_kernel<<<32, 32, 0, stream>>>(jgW + 1 * 16384, W3p, 128, 128, 0);
  pack_w_kernel<<<32, 32, 0, stream>>>(jgW + 2 * 16384, W4p, 128, 128, 0);
  pack_w_kernel<<<128, 32, 0, stream>>>(lstm_Wih, Wihp, 512, 128, 1);
  pack_w_kernel<<<128, 32, 0, stream>>>(lstm_Whh, Whhp, 512, 128, 1);
  pack_w_kernel<<<8, 32, 0, stream>>>(mfe_W, mfeWp, 16, 128, 0);
  pack_w_kernel<<<64, 32, 0, stream>>>(me_W, meWp, 256, 128, 0);
  for (int l = 0; l < 4; ++l)
    pack_w_kernel<<<32, 32, 0, stream>>>(mgW + (size_t)l * 16384,
                                         mgWp + (size_t)l * 16384, 128, 128, 0);

  // Stage graphs: S*B*T = 32768 graphs; graph index = ((s*64+b)*64+t).
  job_enc_kernel<<<32768, 128, 0, stream>>>(j_stages_f, j_stages_a, W1p, W2p,
                                            W3p, W4p, jga, stage_emb);
  // Machine-side job graphs: B*M = 512 graphs.
  job_enc_kernel<<<512, 128, 0, stream>>>(j_machine_f, j_machine_a, W1p, W2p,
                                          W3p, W4p, jga, jmbuf);
  // LSTM over T=64 for the 512 stage sequences.
  lstm_kernel<<<32, 128, 0, stream>>>(stage_emb, Wihp, Whhp, lstm_b, jencbuf);
  // Final machine graph network -> [B,H] output.
  machine_kernel<<<64, 128, 0, stream>>>(machine_f, machine_a, mfeWp, mfe_b,
                                         n2_g, n2_b, meWp, me_b, n3_g, n3_b,
                                         mgWp, mga, mng, mnb, jencbuf, jmbuf,
                                         (float*)d_out);
}
